// DANClassifier_78451872629311
// MI455X (gfx1250) — matrix operand order repaired and checked by reading, not compiled, not run on hardware
//
#include <hip/hip_runtime.h>
#include <hip/hip_bf16.h>

#define VOCAB 400000
#define DIM   300
#define HID   32
#define OUT   2
#define BATCH 16384
#define SEQ   50

#define LDS_PITCH 304   // DIM padded to a multiple of 16 bf16 (32B fragments)

typedef __attribute__((ext_vector_type(16))) __bf16 v16bf;
typedef __attribute__((ext_vector_type(8)))  float  v8f;
typedef __attribute__((ext_vector_type(2)))  float  v2f;

// Convert 8 fp32 (two float4) into bf16 vector elements [base..base+7].
__device__ __forceinline__ void cvt8(v16bf& v, int base, float4 lo, float4 hi) {
    v[base + 0] = (__bf16)lo.x; v[base + 1] = (__bf16)lo.y;
    v[base + 2] = (__bf16)lo.z; v[base + 3] = (__bf16)lo.w;
    v[base + 4] = (__bf16)hi.x; v[base + 5] = (__bf16)hi.y;
    v[base + 6] = (__bf16)hi.z; v[base + 7] = (__bf16)hi.w;
}

// ---------------------------------------------------------------------------
// Kernel 1: P[v, j] = sum_d emb[v,d] * V_w[j,d]   (no bias; bias added later)
// One wave per 16-row vocab tile; two 16-wide N tiles cover HID=32.
// Bulk K (0..287): 9 chunks of V_WMMA_F32_16X16X32_BF16, A converted on the
// fly from the streaming fp32 emb, B served from a block-shared bf16 LDS
// image of V_w. Tail K (288..299): 3 chunks of V_WMMA_F32_16X16X4_F32 in
// full fp32 — branch-free, always in-bounds.
// Fragment layouts per ISA 7.12.2:
//   A 16x32 bf16 : row = lane&15; lanes<16 hold K {0..7,16..23}, lanes>=16
//                  hold K {8..15,24..31}.
//   B 32x16 bf16 : col = lane&15; lanes<16 hold K {0..7,16..23}, lanes>=16 K {8..15,24..31}.
//   A 16x4  f32  : row = lane&15; lanes<16 K {0,1}, lanes>=16 K {2,3}.
//   B 4x16  f32  : col = lane&15; lanes<16 K {0,1}, lanes>=16 K {2,3}.
//   C/D 16x16 f32: VGPR r -> M = r + (lane>=16 ? 8 : 0), N = lane&15.
// ---------------------------------------------------------------------------
__global__ __launch_bounds__(256) void project_kernel(
    const float* __restrict__ emb, const float* __restrict__ Vw,
    float* __restrict__ P)
{
    __shared__ __align__(64) __bf16 VwLds[HID][LDS_PITCH];

    // ---- Stage V_w as bf16 into LDS (once per block), zero-padded to 304.
    for (int idx = threadIdx.x; idx < HID * (LDS_PITCH / 2); idx += 256) {
        const int row = idx / (LDS_PITCH / 2);
        const int k   = (idx % (LDS_PITCH / 2)) * 2;
        // Address-clamped loads (always in-bounds), value-selected to zero.
        const float f0 = Vw[row * DIM + ((k     < DIM) ? k     : 0)];
        const float f1 = Vw[row * DIM + ((k + 1 < DIM) ? k + 1 : 0)];
        VwLds[row][k]     = (k     < DIM) ? (__bf16)f0 : (__bf16)0.0f;
        VwLds[row][k + 1] = (k + 1 < DIM) ? (__bf16)f1 : (__bf16)0.0f;
    }
    __syncthreads();

    const int wave = (blockIdx.x * blockDim.x + threadIdx.x) >> 5; // M tile id
    const int lane = threadIdx.x & 31;
    const int half = lane >> 4;          // 0 or 1
    const int mrow = wave * 16 + (lane & 15);
    const int kgrp = half * 8;           // A K-group offset (bf16): 0 or 8
    const int kb   = half * 8;           // B K-group offset (bf16): 0 or 8
    const int r2   = half * 2;           // K offset (f32 tail): 0 or 2
    const int ncol = lane & 15;

    const float* __restrict__ arow  = emb + (long long)mrow * DIM;
    const float* __restrict__ b0row = Vw  + (long long)(ncol)      * DIM;
    const float* __restrict__ b1row = Vw  + (long long)(ncol + 16) * DIM;

    v8f acc0 = {};
    v8f acc1 = {};

    // 9 full K=32 chunks: max A index = 256+16+8+7 = 287 < 300, all in-bounds.
    for (int kc = 0; kc < 9; ++kc) {
        const int k0 = kc * 32;
        const float4* pa0 = (const float4*)(arow + k0 + kgrp);
        const float4* pa1 = (const float4*)(arow + k0 + 16 + kgrp);
        float4 a00 = pa0[0], a01 = pa0[1];
        float4 a10 = pa1[0], a11 = pa1[1];
        v16bf a;
        cvt8(a, 0, a00, a01);
        cvt8(a, 8, a10, a11);
        typedef __attribute__((ext_vector_type(8))) __bf16 v8bf;
        v8bf b0l = *(const v8bf*)&VwLds[ncol][k0 + kb];
        v8bf b0h = *(const v8bf*)&VwLds[ncol][k0 + 16 + kb];
        v8bf b1l = *(const v8bf*)&VwLds[ncol + 16][k0 + kb];
        v8bf b1h = *(const v8bf*)&VwLds[ncol + 16][k0 + 16 + kb];
        v16bf b0 = __builtin_shufflevector(b0l, b0h, 0,1,2,3,4,5,6,7,8,9,10,11,12,13,14,15);
        v16bf b1 = __builtin_shufflevector(b1l, b1h, 0,1,2,3,4,5,6,7,8,9,10,11,12,13,14,15);
        acc0 = __builtin_amdgcn_wmma_f32_16x16x32_bf16(false, a, false, b0,
                                                       (short)0, acc0, false, false);
        acc1 = __builtin_amdgcn_wmma_f32_16x16x32_bf16(false, a, false, b1,
                                                       (short)0, acc1, false, false);
    }

    // Tail K = 288..299: three fp32 K=4 WMMAs, branch-free and in-bounds
    // (max index = 296 + 2 + 1 = 299).
#pragma unroll
    for (int kc = 0; kc < 3; ++kc) {
        const int k0 = 288 + kc * 4;
        v2f a2  = *(const v2f*)(arow  + k0 + r2);
        v2f b02 = *(const v2f*)(b0row + k0 + r2);
        v2f b12 = *(const v2f*)(b1row + k0 + r2);
        acc0 = __builtin_amdgcn_wmma_f32_16x16x4_f32(false, a2, false, b02,
                                                     (short)0, acc0, false, false);
        acc1 = __builtin_amdgcn_wmma_f32_16x16x4_f32(false, a2, false, b12,
                                                     (short)0, acc1, false, false);
    }

    // Store D tiles: lanes 0-15 of VGPR r write row (wave*16 + r), 64B coalesced.
    const int mbase = wave * 16 + (half << 3);
#pragma unroll
    for (int r = 0; r < 8; ++r) {
        float* prow = P + (long long)(mbase + r) * HID;
        prow[ncol]      = acc0[r];
        prow[ncol + 16] = acc1[r];
    }
}

// ---------------------------------------------------------------------------
// Kernel 2: per batch row, mean-pool projected rows + bias + ReLU + head GEMM.
// One wave per batch row; lane j owns hidden unit j. Logits land in d_out.
// ---------------------------------------------------------------------------
__global__ __launch_bounds__(256) void pool_head_kernel(
    const int* __restrict__ tokens, const float* __restrict__ P,
    const float* __restrict__ Vb, const float* __restrict__ Ww,
    const float* __restrict__ Wb, float* __restrict__ logits)
{
    const int b    = (blockIdx.x * blockDim.x + threadIdx.x) >> 5;
    const int lane = threadIdx.x & 31;
    if (b >= BATCH) return;

    const int* __restrict__ trow = tokens + b * SEQ;
    float acc = 0.0f;
    for (int s = 0; s < SEQ; ++s) {
        const int t = trow[s];
        acc += P[(long long)t * HID + lane];   // 128B rows, L2-resident
    }
    float h = fmaxf(acc * (1.0f / SEQ) + Vb[lane], 0.0f);

    float l0 = h * Ww[lane];           // W_w row 0
    float l1 = h * Ww[HID + lane];     // W_w row 1
#pragma unroll
    for (int off = 16; off > 0; off >>= 1) {
        l0 += __shfl_xor(l0, off, 32);
        l1 += __shfl_xor(l1, off, 32);
    }
    if (lane == 0) {
        logits[b * OUT + 0] = l0 + Wb[0];
        logits[b * OUT + 1] = l1 + Wb[1];
    }
}

// ---------------------------------------------------------------------------
// Kernel 3: per-column (axis=0) max and log-sum-exp over the batch. 1 block.
// stats[o] = max_o + log(sum_b exp(logit[b,o] - max_o))
// ---------------------------------------------------------------------------
__global__ __launch_bounds__(1024) void stats_kernel(
    const float* __restrict__ logits, float* __restrict__ stats)
{
    __shared__ float r0[1024];
    __shared__ float r1[1024];
    const int tid = threadIdx.x;

    float m0 = -3.402823466e38f, m1 = -3.402823466e38f;
    for (int b = tid; b < BATCH; b += 1024) {
        m0 = fmaxf(m0, logits[b * OUT + 0]);
        m1 = fmaxf(m1, logits[b * OUT + 1]);
    }
    r0[tid] = m0; r1[tid] = m1;
    __syncthreads();
    for (int s = 512; s > 0; s >>= 1) {
        if (tid < s) {
            r0[tid] = fmaxf(r0[tid], r0[tid + s]);
            r1[tid] = fmaxf(r1[tid], r1[tid + s]);
        }
        __syncthreads();
    }
    m0 = r0[0]; m1 = r1[0];
    __syncthreads();

    float s0 = 0.0f, s1 = 0.0f;
    for (int b = tid; b < BATCH; b += 1024) {
        s0 += __expf(logits[b * OUT + 0] - m0);
        s1 += __expf(logits[b * OUT + 1] - m1);
    }
    r0[tid] = s0; r1[tid] = s1;
    __syncthreads();
    for (int s = 512; s > 0; s >>= 1) {
        if (tid < s) {
            r0[tid] += r0[tid + s];
            r1[tid] += r1[tid + s];
        }
        __syncthreads();
    }
    if (tid == 0) {
        stats[0] = m0 + __logf(r0[0]);
        stats[1] = m1 + __logf(r1[0]);
    }
}

// Kernel 4: in-place log-softmax finalize on d_out.
__global__ void lsm_write_kernel(float* __restrict__ out,
                                 const float* __restrict__ stats)
{
    const int i = blockIdx.x * blockDim.x + threadIdx.x;
    if (i < BATCH * OUT) out[i] = out[i] - stats[i & 1];
}

// ---------------------------------------------------------------------------
// Fallback (only if ws can't hold the 51 MB projected table): direct DAN.
// One block per batch row.
// ---------------------------------------------------------------------------
__global__ __launch_bounds__(320) void fallback_kernel(
    const int* __restrict__ tokens, const float* __restrict__ emb,
    const float* __restrict__ Vw, const float* __restrict__ Vb,
    const float* __restrict__ Ww, const float* __restrict__ Wb,
    float* __restrict__ logits)
{
    __shared__ int   toks[SEQ];
    __shared__ float pooled[DIM];
    __shared__ float h[HID];
    const int b = blockIdx.x;
    const int tid = threadIdx.x;

    if (tid < SEQ) toks[tid] = tokens[b * SEQ + tid];
    __syncthreads();
    if (tid < DIM) {
        float acc = 0.0f;
        for (int s = 0; s < SEQ; ++s) acc += emb[(long long)toks[s] * DIM + tid];
        pooled[tid] = acc * (1.0f / SEQ);
    }
    __syncthreads();
    if (tid < HID) {
        float acc = 0.0f;
        for (int d = 0; d < DIM; ++d) acc += pooled[d] * Vw[tid * DIM + d];
        h[tid] = fmaxf(acc + Vb[tid], 0.0f);
    }
    __syncthreads();
    if (tid < OUT) {
        float acc = 0.0f;
        for (int j = 0; j < HID; ++j) acc += h[j] * Ww[tid * HID + j];
        logits[b * OUT + tid] = acc + Wb[tid];
    }
}

extern "C" void kernel_launch(void* const* d_in, const int* in_sizes, int n_in,
                              void* d_out, int out_size, void* d_ws, size_t ws_size,
                              hipStream_t stream)
{
    const int*   tokens = (const int*)d_in[0];
    const float* emb    = (const float*)d_in[1];
    const float* Vw     = (const float*)d_in[2];
    const float* Vb     = (const float*)d_in[3];
    const float* Ww     = (const float*)d_in[4];
    const float* Wb     = (const float*)d_in[5];
    float* out = (float*)d_out;   // doubles as logits scratch (same shape)

    const size_t pBytes = (size_t)VOCAB * HID * sizeof(float);
    char* ws = (char*)d_ws;

    if (ws_size >= pBytes + 64) {
        float* P     = (float*)ws;
        float* stats = (float*)(ws + pBytes);
        // 25000 M-tiles -> one wave each; 8 waves / 256-thread block -> exact grid.
        project_kernel<<<(VOCAB / 16) / 8, 256, 0, stream>>>(emb, Vw, P);
        pool_head_kernel<<<BATCH / 8, 256, 0, stream>>>(tokens, P, Vb, Ww, Wb, out);
        stats_kernel<<<1, 1024, 0, stream>>>(out, stats);
        lsm_write_kernel<<<(BATCH * OUT + 255) / 256, 256, 0, stream>>>(out, stats);
    } else {
        float* stats = (float*)ws;   // 8 bytes
        fallback_kernel<<<BATCH, 320, 0, stream>>>(tokens, emb, Vw, Vb, Ww, Wb, out);
        stats_kernel<<<1, 1024, 0, stream>>>(out, stats);
        lsm_write_kernel<<<(BATCH * OUT + 255) / 256, 256, 0, stream>>>(out, stats);
    }
}
